// GraphormerNodesHeadTransform_52347061404300
// MI455X (gfx1250) — matrix operand order repaired and checked by reading, not compiled, not run on hardware
//
#include <hip/hip_runtime.h>
#include <hip/hip_bf16.h>

typedef __bf16 bf16_t;
typedef __attribute__((ext_vector_type(16))) __bf16 v16bf;
typedef __attribute__((ext_vector_type(8)))  __bf16 v8bf;
typedef __attribute__((ext_vector_type(8)))  float  v8f;

#define B_  32
#define N_  256
#define HS_ 768
#define NH_ 32
#define D_  24
// scale applied to both q and k: (D)^(-1/4) so that q.k carries D^(-1/2)
#define QKSCALE 0.45180100180492246f

__device__ inline v16bf combine16(v8bf lo, v8bf hi) {
  v16bf r;
#pragma unroll
  for (int e = 0; e < 8; ++e) { r[e] = lo[e]; r[e + 8] = hi[e]; }
  return r;
}

__device__ inline v8bf zero8() {
  v8bf z;
#pragma unroll
  for (int e = 0; e < 8; ++e) z[e] = (bf16_t)0.0f;
  return z;
}

// ---------------------------------------------------------------- conversions
__global__ void f32_to_bf16(const float* __restrict__ src,
                            bf16_t* __restrict__ dst, int n) {
  int i = blockIdx.x * blockDim.x + threadIdx.x;
  if (i < n) dst[i] = (bf16_t)src[i];
}

// ------------------------------------------------------------------ WMMA GEMM
// C[m,n] = act(A[m,:] . W[n,:] + bias[n]) * scale ; A: MxK bf16, W: NoutxK bf16
// 2x2 register blocking: one 32x32 output block per wave. Per K-step (32):
// 2 A fragments + 2 B fragments (128 B/lane-group) feed 4 WMMAs -> 32 B/WMMA.
// Per-lane operand layouts per CDNA5 ISA 7.12.2:
//   A: lane l holds row (l&15); elems 0-7 = K[(l>>4)*8 .. +7], elems 8-15 = K[16+(l>>4)*8 ..]
//   B: lane l holds col (l&15) of B = row (l&15) of W; elems 0-7 = K[(l>>4)*8 .. +7], elems 8-15 = K[16+(l>>4)*8 ..] (same as A)
//   D: elem r, lane l -> (M = r + 8*(l>>4), N = l&15)
__global__ void __launch_bounds__(256)
gemm_bf16_wmma(const bf16_t* __restrict__ A, const bf16_t* __restrict__ W,
               const float* __restrict__ bias,
               bf16_t* __restrict__ outb, float* __restrict__ outf,
               int M, int K, int Nout, float scale, int do_gelu) {
  int gw   = blockIdx.x * 8 + (threadIdx.x >> 5);
  int lane = threadIdx.x & 31;
  int tiles_n = Nout >> 5;
  int tiles_m = M >> 5;
  if (gw >= tiles_m * tiles_n) return;
  int tn = gw % tiles_n, tm = gw / tiles_n;
  int m0 = tm << 5, n0 = tn << 5;
  int half = lane >> 4;
  int l16  = lane & 15;

  const bf16_t* arow0 = A + (size_t)(m0 + l16) * K;
  const bf16_t* arow1 = A + (size_t)(m0 + 16 + l16) * K;
  const bf16_t* wrow0 = W + (size_t)(n0 + l16) * K;
  const bf16_t* wrow1 = W + (size_t)(n0 + 16 + l16) * K;

  v8f acc00, acc01, acc10, acc11;
#pragma unroll
  for (int r = 0; r < 8; ++r) {
    acc00[r] = 0.0f; acc01[r] = 0.0f; acc10[r] = 0.0f; acc11[r] = 0.0f;
  }

  for (int k0 = 0; k0 < K; k0 += 32) {
    int aoff = k0 + half * 8;
    v16bf a0 = combine16(*(const v8bf*)(arow0 + aoff),
                         *(const v8bf*)(arow0 + aoff + 16));
    v16bf a1 = combine16(*(const v8bf*)(arow1 + aoff),
                         *(const v8bf*)(arow1 + aoff + 16));
    int boff = k0 + half * 8;
    v16bf b0 = combine16(*(const v8bf*)(wrow0 + boff),
                         *(const v8bf*)(wrow0 + boff + 16));
    v16bf b1 = combine16(*(const v8bf*)(wrow1 + boff),
                         *(const v8bf*)(wrow1 + boff + 16));

    acc00 = __builtin_amdgcn_wmma_f32_16x16x32_bf16(false, a0, false, b0,
                                                    (short)0, acc00, false, false);
    acc01 = __builtin_amdgcn_wmma_f32_16x16x32_bf16(false, a0, false, b1,
                                                    (short)0, acc01, false, false);
    acc10 = __builtin_amdgcn_wmma_f32_16x16x32_bf16(false, a1, false, b0,
                                                    (short)0, acc10, false, false);
    acc11 = __builtin_amdgcn_wmma_f32_16x16x32_bf16(false, a1, false, b1,
                                                    (short)0, acc11, false, false);
  }

  int nc0 = n0 + l16;
  int nc1 = n0 + 16 + l16;
  float bv0 = bias[nc0];
  float bv1 = bias[nc1];
#pragma unroll
  for (int r = 0; r < 8; ++r) {
    int mr0 = m0 + r + half * 8;
    int mr1 = mr0 + 16;
    float x00 = acc00[r] + bv0;
    float x01 = acc01[r] + bv1;
    float x10 = acc10[r] + bv0;
    float x11 = acc11[r] + bv1;
    if (do_gelu) {
      x00 = 0.5f * x00 * (1.0f + erff(x00 * 0.70710678118654752f));
      x01 = 0.5f * x01 * (1.0f + erff(x01 * 0.70710678118654752f));
      x10 = 0.5f * x10 * (1.0f + erff(x10 * 0.70710678118654752f));
      x11 = 0.5f * x11 * (1.0f + erff(x11 * 0.70710678118654752f));
    }
    x00 *= scale; x01 *= scale; x10 *= scale; x11 *= scale;
    size_t o00 = (size_t)mr0 * Nout + nc0;
    size_t o01 = (size_t)mr0 * Nout + nc1;
    size_t o10 = (size_t)mr1 * Nout + nc0;
    size_t o11 = (size_t)mr1 * Nout + nc1;
    if (outb) {
      outb[o00] = (bf16_t)x00; outb[o01] = (bf16_t)x01;
      outb[o10] = (bf16_t)x10; outb[o11] = (bf16_t)x11;
    } else {
      outf[o00] = x00; outf[o01] = x01;
      outf[o10] = x10; outf[o11] = x11;
    }
  }
}

// ------------------------------------------------ wv_c[b,h,j] = v.Wf_c (per head)
__global__ void compute_wv(const float* __restrict__ vf,
                           const float* __restrict__ Wf1,
                           const float* __restrict__ Wf2,
                           const float* __restrict__ Wf3,
                           float* __restrict__ wv) {
  int t = blockIdx.x * blockDim.x + threadIdx.x;
  if (t >= B_ * NH_ * N_) return;
  int j = t % N_;
  int h = (t / N_) % NH_;
  int b = t / (N_ * NH_);
  const float* vrow = vf + (size_t)(b * N_ + j) * HS_ + h * D_;
  const float* w1 = Wf1 + h * D_;
  const float* w2 = Wf2 + h * D_;
  const float* w3 = Wf3 + h * D_;
  float s0 = 0.f, s1 = 0.f, s2 = 0.f;
#pragma unroll
  for (int d = 0; d < D_; ++d) {
    float x = vrow[d];
    s0 += x * w1[d]; s1 += x * w2[d]; s2 += x * w3[d];
  }
  size_t o = (size_t)t * 3;
  wv[o + 0] = s0; wv[o + 1] = s1; wv[o + 2] = s2;
}

__global__ void init_out(float* __restrict__ out, const float* __restrict__ bf1,
                         const float* __restrict__ bf2, const float* __restrict__ bf3) {
  int i = blockIdx.x * blockDim.x + threadIdx.x;
  if (i >= B_ * N_ * 3) return;
  int c = i % 3;
  out[i] = (c == 0) ? bf1[0] : (c == 1) ? bf2[0] : bf3[0];
}

// --------------------------------------------------------- attention + reduce
// One block = one (b, h, 16-row i-slab). 4 waves x 4 j-tiles of WMMA scores,
// LDS-staged, then per-row softmax fused with sum_j p*delta*wv.
#define SROW 260  // 256 + 4 pad: rows r and r+8 land in different bank groups
__global__ void __launch_bounds__(128)
attention_kernel(const bf16_t* __restrict__ qb, const bf16_t* __restrict__ kb,
                 const float* __restrict__ bias, const float* __restrict__ mask,
                 const float* __restrict__ delta, const float* __restrict__ wv,
                 float* __restrict__ out) {
  __shared__ float S[16 * SROW];
  int tile = blockIdx.x;
  int i0 = (tile & 15) << 4;
  int h  = (tile >> 4) & (NH_ - 1);
  int b  = tile >> 9;          // / (16*NH_)
  int wave = threadIdx.x >> 5;
  int lane = threadIdx.x & 31;
  int half = lane >> 4, l16 = lane & 15;

  // A = Q rows (16 x 32, d=24 zero-padded to 32)
  const bf16_t* qrow = qb + (size_t)(b * N_ + i0 + l16) * HS_ + h * D_;
  v8bf alo = *(const v8bf*)(qrow + half * 8);               // d in [0,16)
  v8bf ahi = (half == 0) ? *(const v8bf*)(qrow + 16)        // d in [16,24)
                         : zero8();                         // d in [24,32) -> 0
  v16bf a = combine16(alo, ahi);

#pragma unroll
  for (int t4 = 0; t4 < 4; ++t4) {
    int j0 = wave * 64 + t4 * 16;
    const bf16_t* krow = kb + (size_t)(b * N_ + j0 + l16) * HS_ + h * D_;
    v8bf blo, bhi;
    if (half == 0) { blo = *(const v8bf*)(krow);      bhi = *(const v8bf*)(krow + 16); }
    else           { blo = *(const v8bf*)(krow + 8);  bhi = zero8(); }
    v16bf bm = combine16(blo, bhi);

    v8f acc;
#pragma unroll
    for (int r = 0; r < 8; ++r) acc[r] = 0.0f;
    acc = __builtin_amdgcn_wmma_f32_16x16x32_bf16(false, a, false, bm,
                                                  (short)0, acc, false, false);
#pragma unroll
    for (int r = 0; r < 8; ++r)
      S[(r + half * 8) * SROW + j0 + l16] = acc[r];
  }
  __syncthreads();

  // each wave owns 4 complete rows
  for (int rr = 0; rr < 4; ++rr) {
    int r  = wave * 4 + rr;
    int ig = i0 + r;
    const float* brow = bias + (((size_t)b * NH_ + h) * N_ + ig) * (size_t)N_;
    const float* mrow = mask + (size_t)b * N_;
    float* srow = S + r * SROW;

    float mx = -3.4e38f;
#pragma unroll
    for (int k = 0; k < 8; ++k) {
      int j = lane + 32 * k;
      float s = srow[j] + brow[j] + (1.0f - mrow[j]) * -100000.0f;
      srow[j] = s;
      mx = fmaxf(mx, s);
    }
#pragma unroll
    for (int m = 16; m >= 1; m >>= 1) mx = fmaxf(mx, __shfl_xor(mx, m, 32));

    float sum = 0.0f;
#pragma unroll
    for (int k = 0; k < 8; ++k) {
      int j = lane + 32 * k;
      float e = __expf(srow[j] - mx);
      srow[j] = e;
      sum += e;
    }
#pragma unroll
    for (int m = 16; m >= 1; m >>= 1) sum += __shfl_xor(sum, m, 32);
    float rinv = 1.0f / sum;

    const float* dp  = delta + (size_t)(b * N_ + ig) * N_ * 3;
    const float* wvp = wv + (size_t)(b * NH_ + h) * N_ * 3;
    float y0 = 0.f, y1 = 0.f, y2 = 0.f;
#pragma unroll
    for (int k = 0; k < 8; ++k) {
      int j = lane + 32 * k;
      float p = srow[j];
      const float* d3 = dp + j * 3;
      const float* w3 = wvp + j * 3;
      y0 += p * d3[0] * w3[0];
      y1 += p * d3[1] * w3[1];
      y2 += p * d3[2] * w3[2];
    }
#pragma unroll
    for (int m = 16; m >= 1; m >>= 1) {
      y0 += __shfl_xor(y0, m, 32);
      y1 += __shfl_xor(y1, m, 32);
      y2 += __shfl_xor(y2, m, 32);
    }
    if (lane == 0) {
      float* op = out + (size_t)(b * N_ + ig) * 3;
      atomicAdd(op + 0, y0 * rinv);
      atomicAdd(op + 1, y1 * rinv);
      atomicAdd(op + 2, y2 * rinv);
    }
  }
}

// ------------------------------------------------------------------- launcher
extern "C" void kernel_launch(void* const* d_in, const int* in_sizes, int n_in,
                              void* d_out, int out_size, void* d_ws, size_t ws_size,
                              hipStream_t stream) {
  (void)in_sizes; (void)n_in; (void)out_size; (void)ws_size;
  const float* hidden = (const float*)d_in[0];
  const float* abias  = (const float*)d_in[1];
  const float* delta  = (const float*)d_in[2];
  const float* mask   = (const float*)d_in[3];
  const float* Wi  = (const float*)d_in[4];
  const float* bi  = (const float*)d_in[5];
  const float* Wq  = (const float*)d_in[6];
  const float* bq  = (const float*)d_in[7];
  const float* Wk  = (const float*)d_in[8];
  const float* bk  = (const float*)d_in[9];
  const float* Wv  = (const float*)d_in[10];
  const float* bv  = (const float*)d_in[11];
  const float* Wf1 = (const float*)d_in[12];
  const float* bf1 = (const float*)d_in[13];
  const float* Wf2 = (const float*)d_in[14];
  const float* bf2 = (const float*)d_in[15];
  const float* Wf3 = (const float*)d_in[16];
  const float* bf3 = (const float*)d_in[17];
  float* out = (float*)d_out;

  char* ws = (char*)d_ws;
  size_t off = 0;
  auto alloc = [&](size_t bytes) -> char* {
    char* p = ws + off;
    off += (bytes + 255) & ~(size_t)255;
    return p;
  };
  const size_t MK = (size_t)B_ * N_ * HS_;  // 6,291,456
  const size_t WK = (size_t)HS_ * HS_;      //   589,824
  bf16_t* hsb = (bf16_t*)alloc(MK * 2);
  bf16_t* wib = (bf16_t*)alloc(WK * 2);
  bf16_t* wqb = (bf16_t*)alloc(WK * 2);
  bf16_t* wkb = (bf16_t*)alloc(WK * 2);
  bf16_t* wvb = (bf16_t*)alloc(WK * 2);
  bf16_t* hb  = (bf16_t*)alloc(MK * 2);
  bf16_t* qbf = (bf16_t*)alloc(MK * 2);
  bf16_t* kbf = (bf16_t*)alloc(MK * 2);
  float*  vf  = (float*)alloc(MK * 4);
  float*  wvv = (float*)alloc((size_t)B_ * NH_ * N_ * 3 * 4);

  f32_to_bf16<<<(int)((MK + 255) / 256), 256, 0, stream>>>(hidden, hsb, (int)MK);
  f32_to_bf16<<<(int)((WK + 255) / 256), 256, 0, stream>>>(Wi, wib, (int)WK);
  f32_to_bf16<<<(int)((WK + 255) / 256), 256, 0, stream>>>(Wq, wqb, (int)WK);
  f32_to_bf16<<<(int)((WK + 255) / 256), 256, 0, stream>>>(Wk, wkb, (int)WK);
  f32_to_bf16<<<(int)((WK + 255) / 256), 256, 0, stream>>>(Wv, wvb, (int)WK);

  int Mtot = B_ * N_;                          // 8192
  int tiles = (Mtot / 32) * (HS_ / 32);        // 6144 waves (32x32 per wave)
  int gblocks = tiles / 8;                     // 8 waves per 256-thread block
  gemm_bf16_wmma<<<gblocks, 256, 0, stream>>>(hsb, wib, bi, hb, nullptr,
                                              Mtot, HS_, HS_, 1.0f, 1);
  gemm_bf16_wmma<<<gblocks, 256, 0, stream>>>(hb, wqb, bq, qbf, nullptr,
                                              Mtot, HS_, HS_, QKSCALE, 0);
  gemm_bf16_wmma<<<gblocks, 256, 0, stream>>>(hb, wkb, bk, kbf, nullptr,
                                              Mtot, HS_, HS_, QKSCALE, 0);
  gemm_bf16_wmma<<<gblocks, 256, 0, stream>>>(hb, wvb, bv, nullptr, vf,
                                              Mtot, HS_, HS_, 1.0f, 0);

  int nwv = B_ * NH_ * N_;
  compute_wv<<<(nwv + 255) / 256, 256, 0, stream>>>(vf, Wf1, Wf2, Wf3, wvv);
  init_out<<<(B_ * N_ * 3 + 255) / 256, 256, 0, stream>>>(out, bf1, bf2, bf3);

  attention_kernel<<<B_ * NH_ * (N_ / 16), 128, 0, stream>>>(
      qbf, kbf, abias, mask, delta, wvv, out);
}
